// HeteroRGCN_LinkPredictor_31396210933902
// MI455X (gfx1250) — matrix operand order repaired and checked by reading, not compiled, not run on hardware
//
#include <hip/hip_runtime.h>
#include <hip/hip_bf16.h>

#define NGENE 50000
#define NPAT  20000
#define HDIM  256

typedef __attribute__((ext_vector_type(16))) __bf16       v16bf;
typedef __attribute__((ext_vector_type(8)))  float        v8f;
typedef __attribute__((ext_vector_type(8)))  unsigned int v8u;

// fp32 -> bf16 with round-to-nearest-even, returned as raw bits
static __device__ __forceinline__ unsigned short f2bf_u(float f) {
  unsigned int u = __float_as_uint(f);
  unsigned int r = u + 0x7fffu + ((u >> 16) & 1u);
  return (unsigned short)(r >> 16);
}

// Convert fp32 weights [nmat][K][N] to bf16 transposed [nmat][N][K]
__global__ void wprep_kernel(const float* __restrict__ W, unsigned short* __restrict__ Wt,
                             int K, int N, int nmat) {
  long long idx = (long long)blockIdx.x * blockDim.x + threadIdx.x;
  long long tot = (long long)K * N * nmat;
  if (idx >= tot) return;
  int per = K * N;
  int m   = (int)(idx / per);
  int rem = (int)(idx % per);
  int k = rem / N, n = rem % N;
  Wt[(long long)m * per + (long long)n * K + k] = f2bf_u(W[idx]);
}

__global__ void zero_kernel(float* __restrict__ p, long long n) {
  long long i  = (long long)blockIdx.x * blockDim.x + threadIdx.x;
  long long st = (long long)gridDim.x * blockDim.x;
  for (; i < n; i += st) p[i] = 0.0f;
}

// scatter-add src features along edges into dst accumulator + per-dst count
// 256 threads handle 4 edges (64 threads/edge, float4 per thread)
__global__ void edge_scatter_kernel(const float* __restrict__ src_feat,
                                    const int* __restrict__ src_idx,
                                    const int* __restrict__ dst_idx,
                                    float* __restrict__ acc, float* __restrict__ cnt, int E) {
  int e = blockIdx.x * 4 + (threadIdx.x >> 6);
  if (e >= E) return;
  int f = (threadIdx.x & 63) * 4;
  int s = src_idx[e], d = dst_idx[e];
  const float4 v = *(const float4*)(src_feat + (long long)s * HDIM + f);
  float* dp = acc + (long long)d * HDIM + f;
  atomicAdd(dp + 0, v.x);
  atomicAdd(dp + 1, v.y);
  atomicAdd(dp + 2, v.z);
  atomicAdd(dp + 3, v.w);
  if ((threadIdx.x & 63) == 0) atomicAdd(cnt + d, 1.0f);
}

// out[M,256] = sum_s maybe_mean(A_s)[M,K] @ W_s[K,256]  (+bias0+bias1) * scale, opt ReLU
// W_s given as bf16 transposed [256,K]. C_s (optional) = per-row counts for mean.
// 512 threads = 16 waves; wave w owns output columns [16w,16w+16); block owns 16 rows.
template <int NA, int K>
__global__ __launch_bounds__(512)
void sage_gemm(const float* __restrict__ A0, const float* __restrict__ C0, const unsigned short* __restrict__ W0,
               const float* __restrict__ A1, const float* __restrict__ C1, const unsigned short* __restrict__ W1,
               const float* __restrict__ A2, const float* __restrict__ C2, const unsigned short* __restrict__ W2,
               const float* __restrict__ A3, const float* __restrict__ C3, const unsigned short* __restrict__ W3,
               const float* __restrict__ bias0, const float* __restrict__ bias1,
               float scale, int do_relu, float* __restrict__ out, int M) {
  __shared__ __align__(32) unsigned short smA[NA * 16 * K];
  const float* As[4]          = {A0, A1, A2, A3};
  const float* Cs[4]          = {C0, C1, C2, C3};
  const unsigned short* Ws[4] = {W0, W1, W2, W3};
  const int rowBase = blockIdx.x * 16;

  // stage NA strips of A (fp32 -> bf16) into LDS, applying segment-mean divide
  for (int idx = threadIdx.x; idx < NA * 16 * K; idx += 512) {
    int s   = idx / (16 * K);
    int rem = idx - s * 16 * K;
    int r = rem / K;
    int k = rem - r * K;
    int row = rowBase + r;
    float v = 0.0f;
    if (row < M) {
      v = As[s][(long long)row * K + k];
      if (Cs[s]) v *= 1.0f / fmaxf(Cs[s][row], 1.0f);
    }
    smA[idx] = f2bf_u(v);
  }
  __syncthreads();

  const int wave  = threadIdx.x >> 5;
  const int lane  = threadIdx.x & 31;
  const int m16   = lane & 15;
  const int g     = lane >> 4;
  const int n     = wave * 16 + m16;

  v8f c = {0.f, 0.f, 0.f, 0.f, 0.f, 0.f, 0.f, 0.f};

  for (int kb = 0; kb < K; kb += 32) {
#pragma unroll
    for (int s = 0; s < NA; ++s) {
      // A fragment (16x32 bf16): lane m16, group g; K runs [kb+16g, +16)
      const unsigned short* arow = &smA[s * 16 * K + m16 * K + kb + g * 16];
      uint4 lo = *(const uint4*)(arow);
      uint4 hi = *(const uint4*)(arow + 8);
      v8u au;
      au[0] = lo.x; au[1] = lo.y; au[2] = lo.z; au[3] = lo.w;
      au[4] = hi.x; au[5] = hi.y; au[6] = hi.z; au[7] = hi.w;
      v16bf a = __builtin_bit_cast(v16bf, au);
      // B fragment (32x16 bf16): column n, K = kb + 16g + e  (contiguous in W^T[N,K])
      v8u bu = *(const v8u*)(Ws[s] + (long long)n * K + kb + g * 16);
      v16bf b = __builtin_bit_cast(v16bf, bu);
      c = __builtin_amdgcn_wmma_f32_16x16x32_bf16(false, a, false, b, (short)0, c, false, false);
    }
  }

  float bsum = 0.0f;
  if (bias0) bsum += bias0[n];
  if (bias1) bsum += bias1[n];
#pragma unroll
  for (int i = 0; i < 8; ++i) {  // C/D layout: VGPR i holds row 8g+i, col n
    int row = rowBase + g * 8 + i;
    if (row < M) {
      float v = (c[i] + bsum) * scale;
      if (do_relu) v = fmaxf(v, 0.0f);
      out[(long long)row * HDIM + n] = v;
    }
  }
}

extern "C" void kernel_launch(void* const* d_in, const int* in_sizes, int n_in,
                              void* d_out, int out_size, void* d_ws, size_t ws_size,
                              hipStream_t stream) {
  const float* x_gene       = (const float*)d_in[0];
  const float* x_patient    = (const float*)d_in[1];
  const float* W_in_gene    = (const float*)d_in[2];
  const float* b_in_gene    = (const float*)d_in[3];
  const float* W_in_patient = (const float*)d_in[4];
  const float* b_in_patient = (const float*)d_in[5];
  const float* Wl           = (const float*)d_in[6];   // [2][3][256][256]
  const float* bl           = (const float*)d_in[7];   // [2][3][256]
  const float* Wr           = (const float*)d_in[8];   // [2][3][256][256]
  const float* Wf_gene      = (const float*)d_in[9];
  const float* bf_gene      = (const float*)d_in[10];
  const float* Wf_patient   = (const float*)d_in[11];
  const float* bf_patient   = (const float*)d_in[12];
  const int* ei_gg = (const int*)d_in[13];
  const int* ei_gp = (const int*)d_in[14];
  const int* ei_pg = (const int*)d_in[15];
  const int E_GG = in_sizes[13] / 2;
  const int E_GP = in_sizes[14] / 2;
  const int E_PG = in_sizes[15] / 2;

  char* w = (char*)d_ws;
  auto alloc = [&](size_t bytes) -> char* {
    char* p = w;
    w += (bytes + 255) & ~(size_t)255;
    return p;
  };
  unsigned short* wt_in_g = (unsigned short*)alloc((size_t)256 * 256 * 2);
  unsigned short* wt_in_p = (unsigned short*)alloc((size_t)128 * 256 * 2);
  unsigned short* wt_l    = (unsigned short*)alloc((size_t)6 * 256 * 256 * 2);
  unsigned short* wt_r    = (unsigned short*)alloc((size_t)6 * 256 * 256 * 2);
  unsigned short* wt_f_g  = (unsigned short*)alloc((size_t)256 * 256 * 2);
  unsigned short* wt_f_p  = (unsigned short*)alloc((size_t)256 * 256 * 2);
  float* hg_a   = (float*)alloc((size_t)NGENE * 256 * 4);
  float* hg_b   = (float*)alloc((size_t)NGENE * 256 * 4);
  float* hp_a   = (float*)alloc((size_t)NPAT * 256 * 4);
  float* hp_b   = (float*)alloc((size_t)NPAT * 256 * 4);
  float* acc_gg = (float*)alloc((size_t)NGENE * 256 * 4);
  float* acc_pg = (float*)alloc((size_t)NGENE * 256 * 4);
  float* acc_gp = (float*)alloc((size_t)NPAT * 256 * 4);
  float* cnt_gg = (float*)alloc((size_t)NGENE * 4);
  float* cnt_pg = (float*)alloc((size_t)NGENE * 4);
  float* cnt_gp = (float*)alloc((size_t)NPAT * 4);
  long long zero_n = ((long long)(w - (char*)acc_gg)) / 4;  // acc_* + cnt_* region

  // --- weight prep: fp32 [K,N] -> bf16 [N,K] ---
  auto wprep = [&](const float* src, unsigned short* dst, int K, int N, int nm) {
    long long tot = (long long)K * N * nm;
    int blocks = (int)((tot + 255) / 256);
    wprep_kernel<<<blocks, 256, 0, stream>>>(src, dst, K, N, nm);
  };
  wprep(W_in_gene, wt_in_g, 256, 256, 1);
  wprep(W_in_patient, wt_in_p, 128, 256, 1);
  wprep(Wl, wt_l, 256, 256, 6);
  wprep(Wr, wt_r, 256, 256, 6);
  wprep(Wf_gene, wt_f_g, 256, 256, 1);
  wprep(Wf_patient, wt_f_p, 256, 256, 1);

  // --- input projections + relu ---
  sage_gemm<1, 256><<<(NGENE + 15) / 16, 512, 0, stream>>>(
      x_gene, nullptr, wt_in_g, nullptr, nullptr, nullptr, nullptr, nullptr, nullptr,
      nullptr, nullptr, nullptr, b_in_gene, nullptr, 1.0f, 1, hg_a, NGENE);
  sage_gemm<1, 128><<<(NPAT + 15) / 16, 512, 0, stream>>>(
      x_patient, nullptr, wt_in_p, nullptr, nullptr, nullptr, nullptr, nullptr, nullptr,
      nullptr, nullptr, nullptr, b_in_patient, nullptr, 1.0f, 1, hp_a, NPAT);

  float* hg_in = hg_a; float* hg_out = hg_b;
  float* hp_in = hp_a; float* hp_out = hp_b;
  for (int l = 0; l < 2; ++l) {
    zero_kernel<<<2048, 256, 0, stream>>>(acc_gg, zero_n);
    edge_scatter_kernel<<<(E_GG + 3) / 4, 256, 0, stream>>>(hg_in, ei_gg, ei_gg + E_GG, acc_gg, cnt_gg, E_GG);
    edge_scatter_kernel<<<(E_GP + 3) / 4, 256, 0, stream>>>(hg_in, ei_gp, ei_gp + E_GP, acc_gp, cnt_gp, E_GP);
    edge_scatter_kernel<<<(E_PG + 3) / 4, 256, 0, stream>>>(hp_in, ei_pg, ei_pg + E_PG, acc_pg, cnt_pg, E_PG);

    const unsigned short* wl0 = wt_l + (size_t)(l * 3 + 0) * 65536;
    const unsigned short* wl1 = wt_l + (size_t)(l * 3 + 1) * 65536;
    const unsigned short* wl2 = wt_l + (size_t)(l * 3 + 2) * 65536;
    const unsigned short* wr0 = wt_r + (size_t)(l * 3 + 0) * 65536;
    const unsigned short* wr1 = wt_r + (size_t)(l * 3 + 1) * 65536;
    const unsigned short* wr2 = wt_r + (size_t)(l * 3 + 2) * 65536;
    const float* bl0 = bl + (l * 3 + 0) * 256;
    const float* bl1 = bl + (l * 3 + 1) * 256;
    const float* bl2 = bl + (l * 3 + 2) * 256;

    // hg' = relu(0.5*(mean_gg@Wl0 + hg@Wr0 + bl0 + mean_pg@Wl2 + hg@Wr2 + bl2))
    sage_gemm<4, 256><<<(NGENE + 15) / 16, 512, 0, stream>>>(
        acc_gg, cnt_gg, wl0, hg_in, nullptr, wr0, acc_pg, cnt_pg, wl2, hg_in, nullptr, wr2,
        bl0, bl2, 0.5f, 1, hg_out, NGENE);
    // hp' = relu(mean_gp@Wl1 + hp@Wr1 + bl1)
    sage_gemm<2, 256><<<(NPAT + 15) / 16, 512, 0, stream>>>(
        acc_gp, cnt_gp, wl1, hp_in, nullptr, wr1, nullptr, nullptr, nullptr,
        nullptr, nullptr, nullptr, bl1, nullptr, 1.0f, 1, hp_out, NPAT);

    float* t = hg_in; hg_in = hg_out; hg_out = t;
    t = hp_in; hp_in = hp_out; hp_out = t;
  }

  // --- final projections (no relu) straight into d_out ---
  float* out_gene    = (float*)d_out;
  float* out_patient = out_gene + (size_t)NGENE * 256;
  sage_gemm<1, 256><<<(NGENE + 15) / 16, 512, 0, stream>>>(
      hg_in, nullptr, wt_f_g, nullptr, nullptr, nullptr, nullptr, nullptr, nullptr,
      nullptr, nullptr, nullptr, bf_gene, nullptr, 1.0f, 0, out_gene, NGENE);
  sage_gemm<1, 256><<<(NPAT + 15) / 16, 512, 0, stream>>>(
      hp_in, nullptr, wt_f_p, nullptr, nullptr, nullptr, nullptr, nullptr, nullptr,
      nullptr, nullptr, nullptr, bf_patient, nullptr, 1.0f, 0, out_patient, NPAT);
}
